// MoE_55018531061955
// MI455X (gfx1250) — matrix operand order repaired and checked by reading, not compiled, not run on hardware
//
#include <hip/hip_runtime.h>
#include <hip/hip_bf16.h>

// ---------------------------------------------------------------------------
// MoE (T=256, H=1024, I=4096, E=8, top-2) for gfx1250.
// fp32 weights streamed once from HBM (384MB -> ~16.5us floor @23.3TB/s),
// converted to bf16 on the fly, GEMMs via v_wmma_f32_16x16x32_bf16 (wave32).
// Expert segments padded to 128 rows => branch-free K-loop (EXEC all-1s).
// Ping-pong LDS staging (1 barrier per K-step) + register prefetch.
// Conversions use native __bf16 casts (fptrunc RNE) so ISel can pick the
// hardware cvt instructions instead of integer bit-twiddling.
// ---------------------------------------------------------------------------

#define T_TOK 256
#define H_DIM 1024
#define I_DIM 4096
#define N_EXP 8
#define SLOT_CAP 1536         // sum of 128-padded per-expert counts <= 1408
#define STRB 56               // LDS row stride (elems); 112B rows, 16B aligned

typedef __attribute__((ext_vector_type(16))) __bf16 v16bf;
typedef __attribute__((ext_vector_type(2)))  __bf16 v2bf;
typedef __attribute__((ext_vector_type(8)))  float  v8f;

union Frag {
    v16bf v;
    uint4 q[2];
};

// Pack two fp32 -> two bf16 in one dword (lo = first, hi = second).
// Native fptrunc (RNE) lets the backend select hardware cvt ops.
__device__ __forceinline__ unsigned pack_bf16x2(float lo, float hi) {
    v2bf v;
    v.x = (__bf16)lo;
    v.y = (__bf16)hi;
    unsigned u; __builtin_memcpy(&u, &v, 4);
    return u;
}

// Stage a [2 x 4] (k-pair x 4 columns) fp32 patch as packed bf16 pairs into
// the transposed [n][k] LDS tile. kp2 = 2*kpair, nb = column base.
__device__ __forceinline__ void stage_tile(unsigned short* lds,
                                           float4 ra, float4 rb,
                                           int kp2, int nb) {
    const float la[4] = {ra.x, ra.y, ra.z, ra.w};
    const float lb[4] = {rb.x, rb.y, rb.z, rb.w};
#pragma unroll
    for (int j = 0; j < 4; ++j)
        *(unsigned*)&lds[(nb + j) * STRB + kp2] = pack_bf16x2(la[j], lb[j]);
}

// ws header (ints): [0..7]=cnt  [8..15]=cnt_pad128  [16..23]=seg_off [24]=total
#define WS_PAIR_E    128
#define WS_PAIR_SLOT 2176
#define WS_PAIR_W    4224
#define WS_SLOT_TOK  6272
#define WS_XG        12288
#define WS_HBUF      (WS_XG + SLOT_CAP * H_DIM * 2)          // +3MB
#define WS_YBUF      (WS_HBUF + SLOT_CAP * I_DIM * 2)        // +12.6MB
// total ws usage: WS_YBUF + SLOT_CAP*H_DIM*4 ~= 22.0 MB

// --------------------------- kernel A0: init -------------------------------
__global__ void moe_init(int* hdr, int* slot_token) {
    const int tid = threadIdx.x;
    if (tid < 32) hdr[tid] = 0;
    for (int s = tid; s < SLOT_CAP; s += 256) slot_token[s] = -1;
}

// --------------------------- kernel A1: router -----------------------------
// grid 8 x 256 threads; 8 lanes per token, shfl-xor reduction, softmax+top2.
__global__ void moe_router(const float* __restrict__ hs,
                           const float* __restrict__ gw,
                           int* hdr, int* pair_e, int* pair_rank,
                           float* pair_w) {
    const int tid  = threadIdx.x;
    const int lane = tid & 31;
    const int wave = tid >> 5;
    const int t    = blockIdx.x * 32 + wave * 4 + (lane >> 3);
    const int sub  = lane & 7;

    float l[N_EXP];
#pragma unroll
    for (int e = 0; e < N_EXP; ++e) l[e] = 0.f;

    const float* xrow = hs + (size_t)t * H_DIM;
    for (int h = sub * 128; h < sub * 128 + 128; ++h) {
        const float x = xrow[h];
        const float4* g = (const float4*)(gw + h * N_EXP);
        const float4 g0 = g[0], g1 = g[1];
        l[0] += x * g0.x; l[1] += x * g0.y; l[2] += x * g0.z; l[3] += x * g0.w;
        l[4] += x * g1.x; l[5] += x * g1.y; l[6] += x * g1.z; l[7] += x * g1.w;
    }
#pragma unroll
    for (int m = 1; m < 8; m <<= 1)
#pragma unroll
        for (int e = 0; e < N_EXP; ++e) l[e] += __shfl_xor(l[e], m, 32);

    if (sub == 0) {
        float mx = l[0];
#pragma unroll
        for (int e = 1; e < N_EXP; ++e) mx = fmaxf(mx, l[e]);
        float p[N_EXP];
#pragma unroll
        for (int e = 0; e < N_EXP; ++e) p[e] = __expf(l[e] - mx);
        // top-2 (lowest index wins ties, like jax top_k); renorm by pair sum
        int i0 = 0; float b0 = p[0];
#pragma unroll
        for (int e = 1; e < N_EXP; ++e) if (p[e] > b0) { b0 = p[e]; i0 = e; }
        int i1 = -1; float b1 = -1.f;
#pragma unroll
        for (int e = 0; e < N_EXP; ++e)
            if (e != i0 && p[e] > b1) { b1 = p[e]; i1 = e; }
        const float inv = 1.f / (b0 + b1);
        const int r0 = atomicAdd(&hdr[i0], 1);
        const int r1 = atomicAdd(&hdr[i1], 1);
        pair_e[2 * t]     = i0; pair_rank[2 * t]     = r0; pair_w[2 * t]     = b0 * inv;
        pair_e[2 * t + 1] = i1; pair_rank[2 * t + 1] = r1; pair_w[2 * t + 1] = b1 * inv;
    }
}

// ------------------- kernel A1b: pad(128) + prefix -------------------------
__global__ void moe_prefix(int* hdr) {
    if (threadIdx.x == 0) {
        int off = 0;
        for (int e = 0; e < N_EXP; ++e) {
            const int cp = (hdr[e] + 127) & ~127;   // pad to full pass
            hdr[8 + e]  = cp;
            hdr[16 + e] = off;
            off += cp;
        }
        hdr[24] = off;
    }
}

// ------------------------ kernel A1c: slot assignment ----------------------
__global__ void moe_slots(const int* __restrict__ hdr,
                          const int* __restrict__ pair_e,
                          int* pair_slot, int* slot_token) {
    const int p = blockIdx.x * 256 + threadIdx.x;   // 0..511
    const int e = pair_e[p];
    const int slot = hdr[16 + e] + pair_slot[p];    // holds rank on entry
    pair_slot[p] = slot;
    slot_token[slot] = p >> 1;
}

// ------------------- kernel A2: gather X, fp32 -> bf16 ---------------------
__global__ void moe_gather(const float* __restrict__ hs,
                           const int* __restrict__ slot_token,
                           unsigned short* __restrict__ Xg) {
    const int s   = blockIdx.x;
    const int tok = slot_token[s];
    const int j   = threadIdx.x * 4;
    float4 v = make_float4(0.f, 0.f, 0.f, 0.f);
    if (tok >= 0) v = *(const float4*)(hs + (size_t)tok * H_DIM + j);
    uint2 o;
    o.x = pack_bf16x2(v.x, v.y);
    o.y = pack_bf16x2(v.z, v.w);
    *(uint2*)(Xg + (size_t)s * H_DIM + j) = o;
}

// --------------- kernel B: h = silu(X@W1) * (X@W3)  (bf16 WMMA) ------------
// grid (I/64, E), 256 thr = 8 waves: 2 M-wavegroups x 4 N-wavegroups.
// Segment rows are a multiple of 128 => branch-free K-loop.
__global__ __launch_bounds__(256) void moe_gemm1(
        const float* __restrict__ w1s, const float* __restrict__ w3s,
        const unsigned short* __restrict__ Xg,
        unsigned short* __restrict__ hbuf, const int* __restrict__ hdr) {
    __shared__ unsigned short ldsW1[2][64 * STRB];
    __shared__ unsigned short ldsW3[2][64 * STRB];
    const int e    = blockIdx.y;
    const int n0   = blockIdx.x * 64;
    const int cntp = hdr[8 + e];
    const int seg  = hdr[16 + e];
    if (cntp == 0) return;
    const int tid  = threadIdx.x;
    const int lane = tid & 31;
    const int wave = __builtin_amdgcn_readfirstlane(tid) >> 5;
    const int wm = wave >> 2, wn = wave & 3;
    const size_t wbase = (size_t)e * (H_DIM * (size_t)I_DIM);
    // staging: thread owns k-pair kp (rows 2kp,2kp+1) x 4 columns nb..nb+3
    const int kp2 = (tid >> 4) * 2;
    const int nb  = (tid & 15) * 4;
    const float* g1base = w1s + wbase + (size_t)kp2 * I_DIM + n0 + nb;
    const float* g3base = w3s + wbase + (size_t)kp2 * I_DIM + n0 + nb;
    const int col = wn * 16 + (lane & 15);
    const int kh  = (lane & 16) >> 1;
    const int kbo = (lane & 16) ? 8 : 0;   // A-frag K sub-offset per lane half

    for (int m0 = 0; m0 < cntp; m0 += 128) {
        v8f acc1[4] = {}, acc3[4] = {};
        // per-tile A row base pointers (constant across K)
        const unsigned short* xrow[4];
#pragma unroll
        for (int i = 0; i < 4; ++i) {
            const int row = seg + m0 + (wm + 2 * i) * 16 + (lane & 15);
            xrow[i] = Xg + (size_t)row * H_DIM + kbo;
        }
        __syncthreads();                 // prior pass done reading buffer 0
        {   // stage tile k0=0 into buffer 0
            const float4* qa1 = (const float4*)g1base;
            const float4* qb1 = (const float4*)(g1base + I_DIM);
            const float4* qa3 = (const float4*)g3base;
            const float4* qb3 = (const float4*)(g3base + I_DIM);
            stage_tile(ldsW1[0], qa1[0], qb1[0], kp2, nb);
            stage_tile(ldsW3[0], qa3[0], qb3[0], kp2, nb);
        }
        int buf = 0;
        for (int k0 = 0; k0 < H_DIM; k0 += 32) {
            __syncthreads();             // tile k0 visible; prev reads drained
            // register-prefetch next tile (overlaps with WMMA below)
            const int kn = (k0 + 32 < H_DIM) ? (k0 + 32) : k0;
            const float* g1 = g1base + (size_t)kn * I_DIM;
            const float* g3 = g3base + (size_t)kn * I_DIM;
            const float4 na1 = *(const float4*)g1;
            const float4 nb1 = *(const float4*)(g1 + I_DIM);
            const float4 na3 = *(const float4*)g3;
            const float4 nb3 = *(const float4*)(g3 + I_DIM);

            // B frags: lane n = col; lanes<16 K {0..7,16..23}, >=16 K {8..15,24..31}
            Frag b1f, b3f;
            {
                const uint4* p1 = (const uint4*)&ldsW1[buf][col * STRB + kh];
                const uint4* p3 = (const uint4*)&ldsW3[buf][col * STRB + kh];
                b1f.q[0] = p1[0]; b1f.q[1] = p1[2];
                b3f.q[0] = p3[0]; b3f.q[1] = p3[2];
            }
            // stage next tile into the other buffer
            stage_tile(ldsW1[buf ^ 1], na1, nb1, kp2, nb);
            stage_tile(ldsW3[buf ^ 1], na3, nb3, kp2, nb);

#pragma unroll
            for (int i = 0; i < 4; ++i) {
                // A frag: lanes<16 K {0..7,16..23}, lanes>=16 K {8..15,24..31}
                Frag af;
                const unsigned short* xp = xrow[i] + k0;
                af.q[0] = *(const uint4*)xp;
                af.q[1] = *(const uint4*)(xp + 16);
                acc1[i] = __builtin_amdgcn_wmma_f32_16x16x32_bf16(
                    false, af.v, false, b1f.v, (short)0, acc1[i], false, false);
                acc3[i] = __builtin_amdgcn_wmma_f32_16x16x32_bf16(
                    false, af.v, false, b3f.v, (short)0, acc3[i], false, false);
            }
            buf ^= 1;
        }
        // SwiGLU + store bf16 h. C/D layout: VGPR r -> M=r (lanes<16) / r+8.
        const int colg  = n0 + col;
        const int rhalf = (lane & 16) ? 8 : 0;
#pragma unroll
        for (int i = 0; i < 4; ++i) {
            const int rowb = seg + m0 + (wm + 2 * i) * 16 + rhalf;
#pragma unroll
            for (int r = 0; r < 8; ++r) {
                const float s1 = acc1[i][r];
                const float s3 = acc3[i][r];
                const float h  = (s1 / (1.f + __expf(-s1))) * s3;
                *(__bf16*)&hbuf[(size_t)(rowb + r) * I_DIM + colg] = (__bf16)h;
            }
        }
    }
}

// ------------------- kernel C: y = h @ W2  (bf16 WMMA) ---------------------
__global__ __launch_bounds__(256) void moe_gemm2(
        const float* __restrict__ w2s,
        const unsigned short* __restrict__ hbuf,
        float* __restrict__ ybuf, const int* __restrict__ hdr) {
    __shared__ unsigned short ldsW[2][64 * STRB];
    const int e    = blockIdx.y;
    const int n0   = blockIdx.x * 64;
    const int cntp = hdr[8 + e];
    const int seg  = hdr[16 + e];
    if (cntp == 0) return;
    const int tid  = threadIdx.x;
    const int lane = tid & 31;
    const int wave = __builtin_amdgcn_readfirstlane(tid) >> 5;
    const int wm = wave >> 2, wn = wave & 3;
    const size_t wbase = (size_t)e * (I_DIM * (size_t)H_DIM);
    const int kp2 = (tid >> 4) * 2;
    const int nb  = (tid & 15) * 4;
    const float* gbase = w2s + wbase + (size_t)kp2 * H_DIM + n0 + nb;
    const int col = wn * 16 + (lane & 15);
    const int kh  = (lane & 16) >> 1;
    const int kbo = (lane & 16) ? 8 : 0;

    for (int m0 = 0; m0 < cntp; m0 += 128) {
        v8f acc[4] = {};
        const unsigned short* hrow[4];
#pragma unroll
        for (int i = 0; i < 4; ++i) {
            const int row = seg + m0 + (wm + 2 * i) * 16 + (lane & 15);
            hrow[i] = hbuf + (size_t)row * I_DIM + kbo;
        }
        __syncthreads();
        {
            const float4* qa = (const float4*)gbase;
            const float4* qb = (const float4*)(gbase + H_DIM);
            stage_tile(ldsW[0], qa[0], qb[0], kp2, nb);
        }
        int buf = 0;
        for (int k0 = 0; k0 < I_DIM; k0 += 32) {
            __syncthreads();
            const int kn = (k0 + 32 < I_DIM) ? (k0 + 32) : k0;
            const float* g = gbase + (size_t)kn * H_DIM;
            const float4 na = *(const float4*)g;
            const float4 nbv = *(const float4*)(g + H_DIM);

            Frag bf;
            {
                const uint4* p = (const uint4*)&ldsW[buf][col * STRB + kh];
                bf.q[0] = p[0]; bf.q[1] = p[2];
            }
            stage_tile(ldsW[buf ^ 1], na, nbv, kp2, nb);

#pragma unroll
            for (int i = 0; i < 4; ++i) {
                Frag af;
                const unsigned short* hp = hrow[i] + k0;
                af.q[0] = *(const uint4*)hp;
                af.q[1] = *(const uint4*)(hp + 16);
                acc[i] = __builtin_amdgcn_wmma_f32_16x16x32_bf16(
                    false, af.v, false, bf.v, (short)0, acc[i], false, false);
            }
            buf ^= 1;
        }
        const int colg  = n0 + col;
        const int rhalf = (lane & 16) ? 8 : 0;
#pragma unroll
        for (int i = 0; i < 4; ++i) {
            const int rowb = seg + m0 + (wm + 2 * i) * 16 + rhalf;
#pragma unroll
            for (int r = 0; r < 8; ++r)
                ybuf[(size_t)(rowb + r) * H_DIM + colg] = acc[i][r];
        }
    }
}

// ---------------- kernel D: out[t] = w0*y[s0] + w1*y[s1] -------------------
__global__ void moe_combine(const float* __restrict__ ybuf,
                            const int* __restrict__ pair_slot,
                            const float* __restrict__ pair_w,
                            float* __restrict__ out) {
    const int gid = blockIdx.x * 256 + threadIdx.x;
    const int t = gid >> 10, h = gid & (H_DIM - 1);
    const int   s0 = pair_slot[2 * t],     s1 = pair_slot[2 * t + 1];
    const float w0 = pair_w[2 * t],        w1 = pair_w[2 * t + 1];
    out[gid] = w0 * ybuf[(size_t)s0 * H_DIM + h] +
               w1 * ybuf[(size_t)s1 * H_DIM + h];
}

// ---------------------------------------------------------------------------
extern "C" void kernel_launch(void* const* d_in, const int* in_sizes, int n_in,
                              void* d_out, int out_size, void* d_ws, size_t ws_size,
                              hipStream_t stream) {
    (void)in_sizes; (void)n_in; (void)out_size; (void)ws_size;
    const float* hs  = (const float*)d_in[0];
    const float* gw  = (const float*)d_in[1];
    const float* w1s = (const float*)d_in[2];
    const float* w2s = (const float*)d_in[3];
    const float* w3s = (const float*)d_in[4];
    float* out = (float*)d_out;

    char* ws = (char*)d_ws;
    int*   hdr        = (int*)ws;
    int*   pair_e     = (int*)(ws + WS_PAIR_E);
    int*   pair_slot  = (int*)(ws + WS_PAIR_SLOT);
    float* pair_w     = (float*)(ws + WS_PAIR_W);
    int*   slot_token = (int*)(ws + WS_SLOT_TOK);
    unsigned short* Xg   = (unsigned short*)(ws + WS_XG);
    unsigned short* hbuf = (unsigned short*)(ws + WS_HBUF);
    float*          ybuf = (float*)(ws + WS_YBUF);

    moe_init   <<<1, 256, 0, stream>>>(hdr, slot_token);
    moe_router <<<8, 256, 0, stream>>>(hs, gw, hdr, pair_e, pair_slot, pair_w);
    moe_prefix <<<1, 32, 0, stream>>>(hdr);
    moe_slots  <<<2, 256, 0, stream>>>(hdr, pair_e, pair_slot, slot_token);
    moe_gather <<<SLOT_CAP, 256, 0, stream>>>(hs, slot_token, Xg);
    moe_gemm1  <<<dim3(I_DIM / 64, N_EXP), 256, 0, stream>>>(w1s, w3s, Xg, hbuf, hdr);
    moe_gemm2  <<<dim3(H_DIM / 64, N_EXP), 256, 0, stream>>>(w2s, hbuf, ybuf, hdr);
    moe_combine<<<(T_TOK * H_DIM) / 256, 256, 0, stream>>>(ybuf, pair_slot, pair_w, out);
}
